// TopoGCN_v3_59828894433565
// MI455X (gfx1250) — matrix operand order repaired and checked by reading, not compiled, not run on hardware
//
#include <hip/hip_runtime.h>
#include <hip/hip_bf16.h>

// ---------------------------------------------------------------------------
// Types for CDNA5 WMMA (wave32, v_wmma_f32_16x16x32_bf16)
// ---------------------------------------------------------------------------
typedef __bf16 bf16_t;
typedef bf16_t v16bf __attribute__((ext_vector_type(16)));
typedef float  v8f   __attribute__((ext_vector_type(8)));
typedef unsigned short u16;
typedef u16 u16x8 __attribute__((ext_vector_type(8)));

__device__ __forceinline__ u16 f2bf(float f) {
  union { float f; unsigned int u; } v; v.f = f;
  unsigned int r = v.u + 0x7FFFu + ((v.u >> 16) & 1u);   // round-to-nearest-even
  return (u16)(r >> 16);
}
__device__ __forceinline__ float bf2f(u16 h) {
  union { unsigned int u; float f; } v; v.u = ((unsigned int)h) << 16;
  return v.f;
}

// CDNA5 async DMA: global -> LDS, 16B per lane, tracked by ASYNCcnt.
__device__ __forceinline__ void async_ld_b128(unsigned lds_off, const u16* ga) {
  asm volatile("global_load_async_to_lds_b128 %0, %1, off"
               :: "v"(lds_off), "v"(ga) : "memory");
}
__device__ __forceinline__ void wait_async0() {
  asm volatile("s_wait_asynccnt 0x0" ::: "memory");
}

// ---------------------------------------------------------------------------
// Tiled bf16 GEMM via WMMA.  C[Mpad x Ncols] = act(A[Mpad x K] * B + bias)
// A:   bf16 row-major, leading dim lda; M pre-padded to a multiple of 128
//      (pad rows hold garbage -> pad C rows garbage, never read downstream).
// Bsw: bf16 *swizzled*: [kt = K/32][col = Ncols][32 k-values]  (k_wconv_sw)
//      -> every WMMA B fragment is 32 contiguous bytes; every k-tile column
//         block is one contiguous chunk in global memory.
// Block: 256 threads = 8 waves; tile M=128, N=128, K-step=32.
// Double-buffered LDS (2x8KB A + 2x8KB B) filled by GLOBAL_LOAD_ASYNC_TO_LDS
// DMA; one barrier + one s_wait_asynccnt per K-step; DMA for tile t+1 overlaps
// the 8 WMMAs of tile t.
// ---------------------------------------------------------------------------
__global__ __launch_bounds__(256)
void k_gemm_bf16(const u16* __restrict__ A, int lda,
                 const u16* __restrict__ Bsw,
                 const float* __restrict__ bias,
                 int Mpad, int K, int Ncols,
                 float* __restrict__ Cf, u16* __restrict__ Cbf, int relu)
{
  __shared__ u16 As[2][128][32];    // 2 x 8 KB
  __shared__ u16 Bs[2][128 * 32];   // 2 x 8 KB

  const int tid  = threadIdx.x;
  const int lane = tid & 31;
  const int wv   = tid >> 5;
  const int half = lane >> 4;
  const int l16  = lane & 15;
  const int mBase = blockIdx.y * 128;
  const int nBase = blockIdx.x * 128;

  union FC  { v8f v; float f[8]; } acc[8];
  union F16 { v16bf v; unsigned int u[8]; } afr, bfr;
#pragma unroll
  for (int c = 0; c < 8; ++c)
#pragma unroll
    for (int r = 0; r < 8; ++r) acc[c].f[r] = 0.f;

  // Per-thread cooperative DMA coordinates (A: 2x16B, B: 2x16B per K-step).
  const int ar0 = (tid * 2) >> 2;
  const int as0 = ((tid * 2) & 3) << 3;
  const int ar1 = (tid * 2 + 1) >> 2;
  const int as1 = ((tid * 2 + 1) & 3) << 3;
  const u16* aptr0 = A + (size_t)(mBase + ar0) * lda + as0;
  const u16* aptr1 = A + (size_t)(mBase + ar1) * lda + as1;
  const size_t bstep = ((size_t)Ncols << 5);
  const u16* bptr = Bsw + (((size_t)nBase) << 5) + tid * 8;

  // Wave-relative LDS byte offsets of this thread's DMA destinations (buf 0);
  // buffer parity adds 8192 bytes (flat-LDS aperture: low 32 bits = offset).
  const unsigned offA0 = (unsigned)(unsigned long long)&As[0][ar0][as0];
  const unsigned offA1 = (unsigned)(unsigned long long)&As[0][ar1][as1];
  const unsigned offB0 = (unsigned)(unsigned long long)&Bs[0][tid * 8];
  const unsigned offB1 = offB0 + 4096u;

  // --- prologue: DMA tile 0 -> LDS buffer 0 ---
  async_ld_b128(offA0, aptr0);
  async_ld_b128(offA1, aptr1);
  async_ld_b128(offB0, bptr);
  async_ld_b128(offB1, bptr + 2048);

  const int nk = K >> 5;
  const int arow = (wv << 4) + l16;

  for (int t = 0; t < nk; ++t) {
    wait_async0();                         // my DMA chunks for tile t landed
    __syncthreads();                       // all waves' chunks landed
    const int cur = t & 1;

    // DMA tile t+1 into the alternate buffer (overlaps the WMMAs below)
    if (t + 1 < nk) {
      const unsigned bsel = (unsigned)((t + 1) & 1) * 8192u;
      async_ld_b128(offA0 + bsel, aptr0 + ((t + 1) << 5));
      async_ld_b128(offA1 + bsel, aptr1 + ((t + 1) << 5));
      const u16* bp = bptr + (size_t)(t + 1) * bstep;
      async_ld_b128(offB0 + bsel, bp);
      async_ld_b128(offB1 + bsel, bp + 2048);
      if (t + 2 < nk)
        __builtin_prefetch(bptr + (size_t)(t + 2) * bstep + tid * 8, 0, 3);
    }

    // --- A fragment (16x32 bf16): one contiguous 32B chunk ---
    {
      const u16* ap = &As[cur][arow][half << 4];          // k = 16*half + 0..15
#pragma unroll
      for (int i = 0; i < 8; ++i) afr.u[i] = ((const unsigned int*)ap)[i];
    }

    // --- 8 WMMAs; B fragment = 32 contiguous bytes in swizzled LDS ---
    const u16* bsbase = &Bs[cur][(l16 << 5) + (half << 4)];
#pragma unroll
    for (int c = 0; c < 8; ++c) {
      const unsigned int* p = (const unsigned int*)(bsbase + (c << 9));
#pragma unroll
      for (int i = 0; i < 8; ++i) bfr.u[i] = p[i];
      acc[c].v = __builtin_amdgcn_wmma_f32_16x16x32_bf16(
          false, afr.v, false, bfr.v, (short)0, acc[c].v, false, false);
    }
  }

  // --- epilogue: C layout VGPR r -> M = r + 8*half, N = lane&15 ---
#pragma unroll
  for (int c = 0; c < 8; ++c) {
    const int col = nBase + (c << 4) + l16;
    const float bv = bias ? bias[col] : 0.f;
#pragma unroll
    for (int r = 0; r < 8; ++r) {
      const int m = mBase + (wv << 4) + r + (half << 3);
      float v = acc[c].f[r] + bv;
      if (relu) v = fmaxf(v, 0.f);
      if (Cf)  Cf[(size_t)m * Ncols + col]  = v;
      if (Cbf) Cbf[(size_t)m * Ncols + col] = f2bf(v);
    }
  }
}

// ---------------------------------------------------------------------------
// Support kernels
// ---------------------------------------------------------------------------
__global__ void k_fill(float* p, int n, float v) {
  int i = blockIdx.x * blockDim.x + threadIdx.x;
  if (i < n) p[i] = v;
}

// f32 (K x 512) -> bf16 swizzled [Kpad/32][512][32] with zero padding.
// dst[((kt*512)+n)*32 + kk] = src[(kt*32+kk)*512 + n]
__global__ void k_wconv_sw(const float* __restrict__ src, u16* __restrict__ dst,
                           int K, int Kpad) {
  int o = blockIdx.x * blockDim.x + threadIdx.x;
  if (o >= Kpad * 512) return;
  int t   = o >> 14;          // / (512*32)
  int rem = o & 16383;
  int n   = rem >> 5;
  int kk  = rem & 31;
  int k   = (t << 5) + kk;
  float v = (k < K) ? src[(size_t)k * 512 + n] : 0.f;
  dst[o] = f2bf(v);
}

// x = concat(feat, goal, info) -> bf16 N x 1056 (pad zeros)
__global__ void k_pack_x(const float* __restrict__ feat, const float* __restrict__ goal,
                         const float* __restrict__ info, u16* __restrict__ X, int N) {
  int i = blockIdx.x * blockDim.x + threadIdx.x;
  if (i >= N * 1056) return;
  int n = i / 1056, d = i - n * 1056;
  float v = 0.f;
  if (d < 512)       v = feat[(size_t)n * 512 + d];
  else if (d < 1024) v = goal[(size_t)n * 512 + (d - 512)];
  else if (d < 1029) v = info[(size_t)n * 5 + (d - 1024)];
  X[i] = f2bf(v);
}

// y = concat(h_bf16, goal, info) -> bf16 N x 1056
__global__ void k_pack_y(const u16* __restrict__ hbf, const float* __restrict__ goal,
                         const float* __restrict__ info, u16* __restrict__ X, int N) {
  int i = blockIdx.x * blockDim.x + threadIdx.x;
  if (i >= N * 1056) return;
  int n = i / 1056, d = i - n * 1056;
  if (d < 512)       { X[i] = hbf[(size_t)n * 512 + d]; return; }
  float v = 0.f;
  if (d < 1024)      v = goal[(size_t)n * 512 + (d - 512)];
  else if (d < 1029) v = info[(size_t)n * 5 + (d - 1024)];
  X[i] = f2bf(v);
}

// per-node: s = g . a1, t = g . a2  (one wave32 per node, float4 loads)
__global__ __launch_bounds__(256)
void k_rowdot2(const float* __restrict__ G, const float* __restrict__ a,
               float* __restrict__ s, float* __restrict__ t, int N) {
  int lane = threadIdx.x & 31;
  int n = blockIdx.x * 8 + (threadIdx.x >> 5);
  if (n >= N) return;
  const float4* g4 = (const float4*)(G + (size_t)n * 512);
  const float4* a1 = (const float4*)a;
  const float4* a2 = (const float4*)(a + 512);
  float s0 = 0.f, s1 = 0.f;
#pragma unroll
  for (int it = 0; it < 4; ++it) {
    int idx = lane + it * 32;
    float4 gv = g4[idx], av = a1[idx], bv = a2[idx];
    s0 += gv.x * av.x + gv.y * av.y + gv.z * av.z + gv.w * av.w;
    s1 += gv.x * bv.x + gv.y * bv.y + gv.z * bv.z + gv.w * bv.w;
  }
#pragma unroll
  for (int off = 16; off > 0; off >>= 1) {
    s0 += __shfl_down(s0, off, 32);
    s1 += __shfl_down(s1, off, 32);
  }
  if (lane == 0) { s[n] = s0; t[n] = s1; }
}

// per-edge: w = exp(-leaky_relu(s[src]+t[dst], 0.2)); denom[src] += w
__global__ void k_edge_w(const int* __restrict__ src, const int* __restrict__ dst,
                         const float* __restrict__ s, const float* __restrict__ t,
                         float* __restrict__ ew, float* __restrict__ denom, int E) {
  int e = blockIdx.x * blockDim.x + threadIdx.x;
  if (e >= E) return;
  int sn = src[e], dn = dst[e];
  float sc = s[sn] + t[dn];
  float lr = sc > 0.f ? sc : 0.2f * sc;
  float w = __expf(-lr);
  ew[e] = w;
  atomicAdd(&denom[sn], w);
}

// per-edge (one wave32 per edge): acc[src] += w * g[dst]   (float4 gather)
__global__ __launch_bounds__(256)
void k_edge_scatter(const int* __restrict__ src, const int* __restrict__ dst,
                    const float* __restrict__ ew, const float* __restrict__ G,
                    float* __restrict__ acc, int E) {
  int lane = threadIdx.x & 31;
  int e = blockIdx.x * 8 + (threadIdx.x >> 5);
  if (e >= E) return;
  float w = ew[e];
  const float4* g4 = (const float4*)(G + (size_t)dst[e] * 512);
  float* o = acc + (size_t)src[e] * 512;
#pragma unroll
  for (int it = 0; it < 4; ++it) {
    int idx = lane + it * 32;
    float4 gv = g4[idx];
    atomicAdd(&o[idx * 4 + 0], w * gv.x);
    atomicAdd(&o[idx * 4 + 1], w * gv.y);
    atomicAdd(&o[idx * 4 + 2], w * gv.z);
    atomicAdd(&o[idx * 4 + 3], w * gv.w);
  }
}

// h = relu(acc / (denom + 1e-10)) -> bf16
__global__ void k_gat_final(const float* __restrict__ acc, const float* __restrict__ denom,
                            u16* __restrict__ hbf, int total) {
  int i = blockIdx.x * blockDim.x + threadIdx.x;
  if (i >= total) return;
  int n = i >> 9;
  float v = acc[i] / (denom[n] + 1e-10f);
  hbf[i] = f2bf(fmaxf(v, 0.f));
}

// out[n] = sigmoid(h[n,:] . V3 + vb3)  (one wave32 per node)
__global__ __launch_bounds__(256)
void k_head(const u16* __restrict__ hbf, const float* __restrict__ V3,
            const float* __restrict__ vb3, float* __restrict__ out, int N) {
  int lane = threadIdx.x & 31;
  int n = blockIdx.x * 8 + (threadIdx.x >> 5);
  if (n >= N) return;
  const u16* h = hbf + (size_t)n * 512;
  float sum = 0.f;
  for (int d = lane; d < 512; d += 32) sum += bf2f(h[d]) * V3[d];
#pragma unroll
  for (int off = 16; off > 0; off >>= 1) sum += __shfl_down(sum, off, 32);
  if (lane == 0) {
    float z = sum + vb3[0];
    out[n] = 1.f / (1.f + __expf(-z));
  }
}

// ---------------------------------------------------------------------------
// Orchestration
// ---------------------------------------------------------------------------
extern "C" void kernel_launch(void* const* d_in, const int* in_sizes, int n_in,
                              void* d_out, int out_size, void* d_ws, size_t ws_size,
                              hipStream_t stream) {
  const float* feat  = (const float*)d_in[0];
  const float* goal  = (const float*)d_in[1];
  const float* info  = (const float*)d_in[2];
  const int*   esrc  = (const int*)d_in[3];
  const int*   edst  = (const int*)d_in[4];
  const float* W1 = (const float*)d_in[5];  const float* b1  = (const float*)d_in[6];
  const float* W2 = (const float*)d_in[7];  const float* b2  = (const float*)d_in[8];
  const float* W3 = (const float*)d_in[9];  const float* b3  = (const float*)d_in[10];
  const float* V1 = (const float*)d_in[11]; const float* vb1 = (const float*)d_in[12];
  const float* V2 = (const float*)d_in[13]; const float* vb2 = (const float*)d_in[14];
  const float* V3 = (const float*)d_in[15]; const float* vb3 = (const float*)d_in[16];
  const float* gatW = (const float*)d_in[17];
  const float* gatA = (const float*)d_in[18];
  float* out = (float*)d_out;

  const int N    = in_sizes[0] / 512;
  const int Mpad = (N + 127) & ~127;       // pad rows -> branch-free GEMM
  const int E    = in_sizes[3];
  const int K1 = 1029, KP = 1056, H = 512;

  // workspace carve-out (activation buffers sized with Mpad rows)
  char* ws = (char*)d_ws;
  size_t off = 0;
  auto alloc = [&](size_t bytes) -> void* {
    void* p = ws + off;
    off = (off + bytes + 255) & ~(size_t)255;
    return p;
  };
  u16*   Xbf   = (u16*)  alloc((size_t)Mpad * KP * 2);
  u16*   Hbf0  = (u16*)  alloc((size_t)Mpad * H * 2);
  u16*   Hbf1  = (u16*)  alloc((size_t)Mpad * H * 2);
  float* G     = (float*)alloc((size_t)Mpad * H * 4);
  float* Acc   = (float*)alloc((size_t)Mpad * H * 4);
  float* Sv    = (float*)alloc((size_t)N * 4);
  float* Tv    = (float*)alloc((size_t)N * 4);
  float* Den   = (float*)alloc((size_t)N * 4);
  float* Ew    = (float*)alloc((size_t)E * 4);
  u16*   W1bf  = (u16*)  alloc((size_t)KP * H * 2);
  u16*   W2bf  = (u16*)  alloc((size_t)H * H * 2);
  u16*   W3bf  = (u16*)  alloc((size_t)H * H * 2);
  u16*   V1bf  = (u16*)  alloc((size_t)KP * H * 2);
  u16*   V2bf  = (u16*)  alloc((size_t)H * H * 2);
  u16*   Wgbf  = (u16*)  alloc((size_t)5 * H * H * 2);
  (void)ws_size; (void)n_in; (void)out_size;

  const int TPB = 256;
  auto blocks = [](long long n, int t) { return (int)((n + t - 1) / t); };

  // --- weight conversion + fragment swizzle (deterministic, capture-safe) ---
  k_wconv_sw<<<blocks((long long)KP * H, TPB), TPB, 0, stream>>>(W1, W1bf, K1, KP);
  k_wconv_sw<<<blocks((long long)H * H, TPB), TPB, 0, stream>>>(W2, W2bf, H, H);
  k_wconv_sw<<<blocks((long long)H * H, TPB), TPB, 0, stream>>>(W3, W3bf, H, H);
  k_wconv_sw<<<blocks((long long)KP * H, TPB), TPB, 0, stream>>>(V1, V1bf, K1, KP);
  k_wconv_sw<<<blocks((long long)H * H, TPB), TPB, 0, stream>>>(V2, V2bf, H, H);
  for (int i = 0; i < 5; ++i)
    k_wconv_sw<<<blocks((long long)H * H, TPB), TPB, 0, stream>>>(
        gatW + (size_t)i * H * H, Wgbf + (size_t)i * H * H, H, H);

  auto gemm = [&](const u16* A, int lda, const u16* Bsw, const float* bias,
                  float* Cf, u16* Cbf, int K, int relu) {
    dim3 g(H / 128, (unsigned)(Mpad / 128));
    k_gemm_bf16<<<g, 256, 0, stream>>>(A, lda, Bsw, bias, Mpad, K, H, Cf, Cbf, relu);
  };

  // --- input MLP ---
  k_pack_x<<<blocks((long long)N * KP, TPB), TPB, 0, stream>>>(feat, goal, info, Xbf, N);
  gemm(Xbf,  KP, W1bf, b1, nullptr, Hbf0, KP, 1);
  gemm(Hbf0, H,  W2bf, b2, nullptr, Hbf1, H,  1);
  gemm(Hbf1, H,  W3bf, b3, nullptr, Hbf0, H,  0);

  // --- 5 GAT layers ---
  for (int i = 0; i < 5; ++i) {
    gemm(Hbf0, H, Wgbf + (size_t)i * H * H, nullptr, G, nullptr, H, 0);
    k_rowdot2<<<blocks(N, 8), 256, 0, stream>>>(G, gatA + (size_t)i * 1024, Sv, Tv, N);
    k_fill<<<blocks(N, TPB), TPB, 0, stream>>>(Den, N, 0.f);
    k_fill<<<blocks((long long)N * H, TPB), TPB, 0, stream>>>(Acc, N * H, 0.f);
    k_edge_w<<<blocks(E, TPB), TPB, 0, stream>>>(esrc, edst, Sv, Tv, Ew, Den, E);
    k_edge_scatter<<<blocks(E, 8), 256, 0, stream>>>(esrc, edst, Ew, G, Acc, E);
    k_gat_final<<<blocks((long long)N * H, TPB), TPB, 0, stream>>>(Acc, Den, Hbf0, N * H);
  }

  // --- output MLP + sigmoid head ---
  k_pack_y<<<blocks((long long)N * KP, TPB), TPB, 0, stream>>>(Hbf0, goal, info, Xbf, N);
  gemm(Xbf,  KP, V1bf, vb1, nullptr, Hbf1, KP, 1);
  gemm(Hbf1, H,  V2bf, vb2, nullptr, Hbf0, H,  1);
  k_head<<<blocks(N, 8), 256, 0, stream>>>(Hbf0, V3, vb3, out, N);
}
